// SelfAttention_41695542509961
// MI455X (gfx1250) — matrix operand order repaired and checked by reading, not compiled, not run on hardware
//
#include <hip/hip_runtime.h>
#include <hip/hip_bf16.h>

typedef __bf16 bf16;
typedef __attribute__((ext_vector_type(16))) __bf16 v16bf;
typedef __attribute__((ext_vector_type(8)))  float v8f;
typedef __attribute__((ext_vector_type(4)))  unsigned int u32x4;
typedef __attribute__((ext_vector_type(4)))  int i32x4;

#define HID 1024
#define N3H 3072
#define BQ  4
#define SEQ 2048
#define MTOT (BQ*SEQ)      // 8192 rows total (B*S)

#define BLK 128            // kernel-1 block tile M and N
#define MBLK 256           // attention GEMM block tile M
#define NBLK 128           // attention GEMM block tile N
#define BK  32             // k-step (matches 16x16x32 bf16 WMMA)
#define LDT 56             // padded LDS row stride in elements (112B: 16B-aligned, conflict-free)

// ---- gfx1250 async global->LDS copies (guarded; falls back to sync copy) ----
#if defined(__gfx1250__) && defined(__has_builtin)
#  if __has_builtin(__builtin_amdgcn_global_load_async_to_lds_b128) && \
      __has_builtin(__builtin_amdgcn_s_wait_asynccnt)
#    define USE_ASYNC_LDS 1
#  endif
#endif
#ifndef USE_ASYNC_LDS
#  define USE_ASYNC_LDS 0
#endif

__device__ __forceinline__ void async_cp16(void* lds, const void* g) {
#if USE_ASYNC_LDS
  __builtin_amdgcn_global_load_async_to_lds_b128(
      (__attribute__((address_space(1))) i32x4*)g,
      (__attribute__((address_space(3))) i32x4*)lds, 0, 0);
#else
  *(u32x4*)lds = *(const u32x4*)g;
#endif
}
__device__ __forceinline__ void async_wait() {
#if USE_ASYNC_LDS
  __builtin_amdgcn_s_wait_asynccnt(0);
#endif
}

union Frag { v16bf v; u32x4 q[2]; };

__device__ __forceinline__ bf16 f2bf(float x) { return (bf16)x; }

// A fragment: 16x32 bf16. Lane L<16: row M=L, K in {0..7,16..23}; L>=16: row M=L-16, K in {8..15,24..31}.
__device__ __forceinline__ Frag load_a_frag(const bf16 tile[][LDT], int row0, int lane) {
  Frag f;
  int r  = row0 + (lane & 15);
  int kh = ((lane >> 4) & 1) * 8;
  const bf16* p = &tile[r][0];
  f.q[0] = *(const u32x4*)(p + kh);
  f.q[1] = *(const u32x4*)(p + 16 + kh);
  return f;
}

// B fragment: 32x16 bf16, tile stored column-major (tile[n][k] = B[k][n]).
// Lane L<16: col N=L, K in {0..7,16..23}; L>=16: col N=L-16, K in {8..15,24..31}.
__device__ __forceinline__ Frag load_b_frag(const bf16 tile[][LDT], int col0, int lane) {
  Frag f;
  int c  = col0 + (lane & 15);
  int kh = ((lane >> 4) & 1) * 8;
  const bf16* p = &tile[c][0];
  f.q[0] = *(const u32x4*)(p + kh);
  f.q[1] = *(const u32x4*)(p + 16 + kh);
  return f;
}

__device__ __forceinline__ v8f wmma_bf16(const Frag& a, const Frag& b, v8f c) {
  return __builtin_amdgcn_wmma_f32_16x16x32_bf16(false, a.v, false, b.v, (short)0, c, false, false);
}

// -------------------- Kernel 1: fused QKV projection --------------------
// proj = X[8192,1024] * W[1024,3072] + b ; -> Qs(*scale) [r][h], K [r][h], V^T [b][h][s]  (all bf16)
__global__ __launch_bounds__(256) void qkv_proj_41695542509961(
    const float* __restrict__ X, const float* __restrict__ W,
    const float* __restrict__ bias, bf16* __restrict__ Qs,
    bf16* __restrict__ Kd, bf16* __restrict__ Vt) {
  __shared__ __align__(16) bf16 As[2][BLK][LDT];
  __shared__ __align__(16) bf16 Bt[2][BLK][LDT];
  const int tid = threadIdx.x, lane = tid & 31, wave = tid >> 5;
  const int wm = wave >> 1, wn = wave & 1;
  const int m0 = blockIdx.y * BLK, n0 = blockIdx.x * BLK;

  // per-thread staging slots (distinct regs -> loads issue back-to-back)
  int ra[4], ca[4], kb[4], cb[4];
  #pragma unroll
  for (int j = 0; j < 4; ++j) {
    int s = tid + j * 256;
    ra[j] = s >> 3;  ca[j] = (s & 7) * 4;    // A: 128 rows x 8 float4
    kb[j] = s >> 5;  cb[j] = (s & 31) * 4;   // B: 32 k-rows x 32 float4
  }
  float4 xa[4], wb[4];

  auto loadRegs = [&](int k0) {
    #pragma unroll
    for (int j = 0; j < 4; ++j)
      xa[j] = *(const float4*)(X + (size_t)(m0 + ra[j]) * HID + k0 + ca[j]);
    #pragma unroll
    for (int j = 0; j < 4; ++j)
      wb[j] = *(const float4*)(W + (size_t)(k0 + kb[j]) * N3H + n0 + cb[j]);
  };
  auto storeRegs = [&](int buf) {
    #pragma unroll
    for (int j = 0; j < 4; ++j) {
      As[buf][ra[j]][ca[j]+0] = f2bf(xa[j].x); As[buf][ra[j]][ca[j]+1] = f2bf(xa[j].y);
      As[buf][ra[j]][ca[j]+2] = f2bf(xa[j].z); As[buf][ra[j]][ca[j]+3] = f2bf(xa[j].w);
    }
    #pragma unroll
    for (int j = 0; j < 4; ++j) {
      Bt[buf][cb[j]+0][kb[j]] = f2bf(wb[j].x); Bt[buf][cb[j]+1][kb[j]] = f2bf(wb[j].y);
      Bt[buf][cb[j]+2][kb[j]] = f2bf(wb[j].z); Bt[buf][cb[j]+3][kb[j]] = f2bf(wb[j].w);
    }
  };

  v8f acc[2][4] = {};
  auto compute = [&](int buf) {
    Frag af[2], bfr[4];
    #pragma unroll
    for (int tm = 0; tm < 2; ++tm) af[tm] = load_a_frag(As[buf], wm*32 + tm*16, lane);
    #pragma unroll
    for (int tn = 0; tn < 4; ++tn) bfr[tn] = load_b_frag(Bt[buf], wn*64 + tn*16, lane);
    #pragma unroll
    for (int tm = 0; tm < 2; ++tm)
      #pragma unroll
      for (int tn = 0; tn < 4; ++tn)
        acc[tm][tn] = wmma_bf16(af[tm], bfr[tn], acc[tm][tn]);
  };

  loadRegs(0); storeRegs(0); __syncthreads();
  for (int k0 = 0; k0 < HID; k0 += 2*BK) {
    loadRegs(k0 + BK);                 // always valid: HID/BK is even
    compute(0);
    storeRegs(1);
    __syncthreads();
    const bool more = (k0 + 2*BK) < HID;
    if (more) loadRegs(k0 + 2*BK);
    compute(1);
    if (more) storeRegs(0);
    __syncthreads();
  }

  const float qscale = 0.03125f;             // 1024^-0.5
  const int rbase = m0 + wm*32 + ((lane >> 4) ? 8 : 0);
  const int cbase = n0 + wn*64 + (lane & 15);
  #pragma unroll
  for (int tm = 0; tm < 2; ++tm)
    #pragma unroll
    for (int tn = 0; tn < 4; ++tn)
      #pragma unroll
      for (int v = 0; v < 8; ++v) {
        int row = rbase + tm*16 + v;
        int col = cbase + tn*16;
        float val = acc[tm][tn][v] + bias[col];
        if (col < HID) {
          Qs[(size_t)row*HID + col] = f2bf(val * qscale);
        } else if (col < 2*HID) {
          Kd[(size_t)row*HID + (col - HID)] = f2bf(val);
        } else {
          int h = col - 2*HID, bb = row >> 11, s = row & (SEQ - 1);
          Vt[((size_t)bb * HID + h) * SEQ + s] = f2bf(val);   // V transposed: [b][h][s]
        }
      }
}

// -------------------- Kernel 2: scores = Qs * K^T  (256x128 block, 64x64 wave tile) ----------
__global__ __launch_bounds__(256) void attn_scores_41695542509961(
    const bf16* __restrict__ Qs, const bf16* __restrict__ Kd, float* __restrict__ Sc) {
  __shared__ __align__(16) bf16 As[2][MBLK][LDT];
  __shared__ __align__(16) bf16 Bt[2][NBLK][LDT];
  const int tid = threadIdx.x, lane = tid & 31, wave = tid >> 5;
  const int wm = wave >> 1, wn = wave & 1;     // 4 x 2 waves, each 64x64
  const int b = blockIdx.z;
  const bf16* Qb = Qs + (size_t)b * SEQ * HID;
  const bf16* Kb = Kd + (size_t)b * SEQ * HID;
  float* Sb = Sc + (size_t)b * SEQ * SEQ;
  const int m0 = blockIdx.y * MBLK, n0 = blockIdx.x * NBLK;

  auto issueTiles = [&](int k0, int buf) {
    #pragma unroll
    for (int j = 0; j < 4; ++j) {              // A: 256x32 = 1024 16B chunks
      int s = tid + j * 256;
      int r = s >> 2, c8 = (s & 3) * 8;
      async_cp16(&As[buf][r][c8], Qb + (size_t)(m0 + r) * HID + k0 + c8);
    }
    #pragma unroll
    for (int j = 0; j < 2; ++j) {              // B: 128x32 = 512 16B chunks; Bt[n][k] = K[n][k]
      int s = tid + j * 256;
      int r = s >> 2, c8 = (s & 3) * 8;
      async_cp16(&Bt[buf][r][c8], Kb + (size_t)(n0 + r) * HID + k0 + c8);
    }
  };

  v8f acc[4][4] = {};
  auto compute = [&](int buf) {
    Frag af[4], bfr[4];
    #pragma unroll
    for (int tm = 0; tm < 4; ++tm) af[tm] = load_a_frag(As[buf], wm*64 + tm*16, lane);
    #pragma unroll
    for (int tn = 0; tn < 4; ++tn) bfr[tn] = load_b_frag(Bt[buf], wn*64 + tn*16, lane);
    #pragma unroll
    for (int tm = 0; tm < 4; ++tm)
      #pragma unroll
      for (int tn = 0; tn < 4; ++tn)
        acc[tm][tn] = wmma_bf16(af[tm], bfr[tn], acc[tm][tn]);
  };

  issueTiles(0, 0); async_wait(); __syncthreads();
  for (int k0 = 0; k0 < HID; k0 += 2*BK) {
    issueTiles(k0 + BK, 1);                    // always valid: HID/BK even
    compute(0);
    async_wait(); __syncthreads();
    if (k0 + 2*BK < HID) issueTiles(k0 + 2*BK, 0);
    compute(1);
    async_wait(); __syncthreads();
  }

  const int rbase = m0 + wm*64 + ((lane >> 4) ? 8 : 0);
  const int cbase = n0 + wn*64 + (lane & 15);
  #pragma unroll
  for (int tm = 0; tm < 4; ++tm)
    #pragma unroll
    for (int tn = 0; tn < 4; ++tn)
      #pragma unroll
      for (int v = 0; v < 8; ++v)
        Sb[(size_t)(rbase + tm*16 + v) * SEQ + cbase + tn*16] = acc[tm][tn][v];
}

// -------------------- Kernel 3: row softmax (fp32 in, bf16 probs in-place) --------------------
__global__ __launch_bounds__(256) void softmax_41695542509961(
    float* __restrict__ Sc, const float* __restrict__ mask) {
  __shared__ float buf[SEQ];
  __shared__ float red[256];
  const int tid = threadIdx.x;
  const int row = blockIdx.x;                // 0..8191
  const int b = row >> 11;
  float* Srow = Sc + (size_t)row * SEQ;
  float lmax = -3.0e38f;
  #pragma unroll
  for (int j = 0; j < SEQ/256; ++j) {
    int i = tid + j * 256;
    float v = Srow[i] + (1.0f - mask[(size_t)b * SEQ + i]) * -10000.0f;
    buf[i] = v;
    lmax = fmaxf(lmax, v);
  }
  red[tid] = lmax; __syncthreads();
  for (int off = 128; off > 0; off >>= 1) {
    if (tid < off) red[tid] = fmaxf(red[tid], red[tid + off]);
    __syncthreads();
  }
  const float rmax = red[0];
  __syncthreads();
  float lsum = 0.0f;
  #pragma unroll
  for (int j = 0; j < SEQ/256; ++j) {
    int i = tid + j * 256;
    float e = __expf(buf[i] - rmax);
    buf[i] = e; lsum += e;
  }
  red[tid] = lsum; __syncthreads();
  for (int off = 128; off > 0; off >>= 1) {
    if (tid < off) red[tid] += red[tid + off];
    __syncthreads();
  }
  const float inv = 1.0f / red[0];
  bf16* Prow = (bf16*)Srow;                  // bf16 probs over the front half of the fp32 row
  #pragma unroll
  for (int j = 0; j < SEQ/256; ++j) {
    int i = tid + j * 256;
    Prow[i] = f2bf(buf[i] * inv);
  }
}

// -------------------- Kernel 4: out = P * V  (V^T layout; 256x128 block, 64x64 wave tile) -----
__global__ __launch_bounds__(256) void attn_out_41695542509961(
    const bf16* __restrict__ Pbase, const bf16* __restrict__ Vt, float* __restrict__ Out) {
  __shared__ __align__(16) bf16 As[2][MBLK][LDT];
  __shared__ __align__(16) bf16 Bt[2][NBLK][LDT];
  const int tid = threadIdx.x, lane = tid & 31, wave = tid >> 5;
  const int wm = wave >> 1, wn = wave & 1;
  const int b = blockIdx.z;
  const bf16* Pb  = Pbase + (size_t)b * SEQ * (2*SEQ);   // probs row stride = 4096 bf16
  const bf16* Vtb = Vt + (size_t)b * HID * SEQ;          // Vt[h][s]
  float* Ob = Out + (size_t)b * SEQ * HID;
  const int m0 = blockIdx.y * MBLK, n0 = blockIdx.x * NBLK;

  auto issueTiles = [&](int k0, int buf) {
    #pragma unroll
    for (int j = 0; j < 4; ++j) {              // A: probs rows (strided 4096)
      int s = tid + j * 256;
      int r = s >> 2, c8 = (s & 3) * 8;
      async_cp16(&As[buf][r][c8], Pb + (size_t)(m0 + r) * (2*SEQ) + k0 + c8);
    }
    #pragma unroll
    for (int j = 0; j < 2; ++j) {              // B: Bt[h][k] = Vt[h][k]
      int s = tid + j * 256;
      int r = s >> 2, c8 = (s & 3) * 8;
      async_cp16(&Bt[buf][r][c8], Vtb + (size_t)(n0 + r) * SEQ + k0 + c8);
    }
  };

  v8f acc[4][4] = {};
  auto compute = [&](int buf) {
    Frag af[4], bfr[4];
    #pragma unroll
    for (int tm = 0; tm < 4; ++tm) af[tm] = load_a_frag(As[buf], wm*64 + tm*16, lane);
    #pragma unroll
    for (int tn = 0; tn < 4; ++tn) bfr[tn] = load_b_frag(Bt[buf], wn*64 + tn*16, lane);
    #pragma unroll
    for (int tm = 0; tm < 4; ++tm)
      #pragma unroll
      for (int tn = 0; tn < 4; ++tn)
        acc[tm][tn] = wmma_bf16(af[tm], bfr[tn], acc[tm][tn]);
  };

  issueTiles(0, 0); async_wait(); __syncthreads();
  for (int k0 = 0; k0 < SEQ; k0 += 2*BK) {
    issueTiles(k0 + BK, 1);                    // always valid: SEQ/BK even
    compute(0);
    async_wait(); __syncthreads();
    if (k0 + 2*BK < SEQ) issueTiles(k0 + 2*BK, 0);
    compute(1);
    async_wait(); __syncthreads();
  }

  const int rbase = m0 + wm*64 + ((lane >> 4) ? 8 : 0);
  const int cbase = n0 + wn*64 + (lane & 15);
  #pragma unroll
  for (int tm = 0; tm < 4; ++tm)
    #pragma unroll
    for (int tn = 0; tn < 4; ++tn)
      #pragma unroll
      for (int v = 0; v < 8; ++v)
        Ob[(size_t)(rbase + tm*16 + v) * HID + cbase + tn*16] = acc[tm][tn][v];
}

extern "C" void kernel_launch(void* const* d_in, const int* in_sizes, int n_in,
                              void* d_out, int out_size, void* d_ws, size_t ws_size,
                              hipStream_t stream) {
  const float* qkv  = (const float*)d_in[0];   // [4,2048,1024]
  const float* mask = (const float*)d_in[1];   // [4,2048]
  const float* W    = (const float*)d_in[2];   // [1024,3072]
  const float* bvec = (const float*)d_in[3];   // [3072]
  float* out = (float*)d_out;                  // [4,2048,1024]

  char* ws = (char*)d_ws;
  const size_t MB = 1024ull * 1024ull;
  bf16*  Qs     = (bf16*)(ws);                 // 16 MB
  bf16*  Kd     = (bf16*)(ws + 16 * MB);       // 16 MB
  bf16*  Vt     = (bf16*)(ws + 32 * MB);       // 16 MB (V transposed [b][h][s])
  float* scores = (float*)(ws + 48 * MB);      // 64 MB (fp32 scores, reused for bf16 probs)

  dim3 blk(256);
  qkv_proj_41695542509961<<<dim3(N3H / BLK, MTOT / BLK), blk, 0, stream>>>(qkv, W, bvec, Qs, Kd, Vt);
  attn_scores_41695542509961<<<dim3(SEQ / NBLK, SEQ / MBLK, BQ), blk, 0, stream>>>(Qs, Kd, scores);
  softmax_41695542509961<<<dim3(MTOT), blk, 0, stream>>>(scores, mask);
  attn_out_41695542509961<<<dim3(HID / NBLK, SEQ / MBLK, BQ), blk, 0, stream>>>((const bf16*)scores, Vt, out);
}
